// StackedLSTM_37323265803178
// MI455X (gfx1250) — matrix operand order repaired and checked by reading, not compiled, not run on hardware
//
#include <hip/hip_runtime.h>
#include <hip/hip_bf16.h>
#include <math.h>

typedef __attribute__((ext_vector_type(16))) _Float16 v16h;
typedef __attribute__((ext_vector_type(8)))  _Float16 v8h;
typedef __attribute__((ext_vector_type(8)))  float    v8f;

#define T_LEN 4096
#define BATCH 2
#define HID 64
#define G4 256

__device__ __forceinline__ float sigf(float x) { return 1.0f / (1.0f + expf(-x)); }

__device__ __forceinline__ v8h cvt8(const float* __restrict__ p) {
  float4 f0 = *(const float4*)p;
  float4 f1 = *(const float4*)(p + 4);
  v8h r;
  r[0] = (_Float16)f0.x; r[1] = (_Float16)f0.y; r[2] = (_Float16)f0.z; r[3] = (_Float16)f0.w;
  r[4] = (_Float16)f1.x; r[5] = (_Float16)f1.y; r[6] = (_Float16)f1.z; r[7] = (_Float16)f1.w;
  return r;
}

__device__ __forceinline__ v16h cat16(v8h lo, v8h hi) {
  return __builtin_shufflevector(lo, hi, 0, 1, 2, 3, 4, 5, 6, 7, 8, 9, 10, 11, 12, 13, 14, 15);
}

// ---------------------------------------------------------------------------
// Zero-pad [rows,Kin] -> [rows,64]
// ---------------------------------------------------------------------------
__global__ void pad_kernel(const float* __restrict__ src, float* __restrict__ dst,
                           int rows, int Kin)
{
  int i = blockIdx.x * blockDim.x + threadIdx.x;
  if (i >= rows * HID) return;
  int r = i >> 6, k = i & 63;
  dst[i] = (k < Kin) ? src[r * Kin + k] : 0.0f;
}

// ---------------------------------------------------------------------------
// Input-projection GEMM: Y[M,256] = X[M,64] @ W[256,64]^T + bias.
// One wave per (16-row tile) x (64-col group): 4 accumulators reuse one A frag.
// All fragment loads are unconditional 128-bit vector loads (K fixed at 64).
// ---------------------------------------------------------------------------
__global__ __launch_bounds__(128) void proj_gemm_kernel(
    const float* __restrict__ X, const float* __restrict__ W,
    const float* __restrict__ bias, float* __restrict__ Y)
{
  const int lane = threadIdx.x & 31;
  const int wave = blockIdx.x * (blockDim.x >> 5) + (threadIdx.x >> 5);
  const int tile_m = wave >> 2;   // 512 row tiles
  const int ng     = wave & 3;    // 4 col groups of 64

  const int m_a    = tile_m * 16 + (lane & 15);
  const int kb_a   = (lane < 16) ? 0 : 8;
  const int n_lane = lane & 15;
  const int kb_b   = (lane < 16) ? 0 : 8;

  const float* __restrict__ xrow = X + (size_t)m_a * HID;

  v8f acc[4] = {};
#pragma unroll
  for (int kc = 0; kc < 2; ++kc) {
    const int k0 = kc * 32;
    v16h a = cat16(cvt8(xrow + k0 + kb_a), cvt8(xrow + k0 + kb_a + 16));
#pragma unroll
    for (int j = 0; j < 4; ++j) {
      int n = (ng * 4 + j) * 16 + n_lane;
      const float* __restrict__ wrow = W + (size_t)n * HID;
      v16h b = cat16(cvt8(wrow + k0 + kb_b), cvt8(wrow + k0 + kb_b + 16));
      acc[j] = __builtin_amdgcn_wmma_f32_16x16x32_f16(false, a, false, b,
                                                      (short)0, acc[j], false, false);
    }
  }
#pragma unroll
  for (int j = 0; j < 4; ++j) {
    int n = (ng * 4 + j) * 16 + n_lane;
    float bv = bias[n];
#pragma unroll
    for (int v = 0; v < 8; ++v) {
      int m = tile_m * 16 + v + ((lane < 16) ? 0 : 8);
      Y[(size_t)m * G4 + n] = acc[j][v] + bv;
    }
  }
}

// ---------------------------------------------------------------------------
// Persistent LSTM scan for one layer. One workgroup, 8 waves.
// Whh^T B-fragments live in registers for all 4096 steps; h lives in a
// zero-padded 16x64 f16 LDS tile so A-fragment loads are two ds_load_b128.
// ---------------------------------------------------------------------------
__global__ __launch_bounds__(256) void lstm_scan_kernel(
    const float* __restrict__ xg, const float* __restrict__ Whh,
    const float* __restrict__ h0, const float* __restrict__ c0,
    float* __restrict__ out, int T)
{
  __shared__ float    cs[BATCH][HID];
  __shared__ float    gsh[BATCH][G4];
  __shared__ _Float16 hf[16][HID];          // rows >= BATCH stay zero

  const int tid  = threadIdx.x;
  const int lane = tid & 31;
  const int wave = tid >> 5;

  for (int i = tid; i < 16 * HID; i += 256) ((_Float16*)hf)[i] = (_Float16)0.0f;
  if (tid < BATCH * HID) {
    int b = tid >> 6, h = tid & 63;
    hf[b][h] = (_Float16)h0[b * HID + h];
    cs[b][h] = c0[b * HID + h];
  }

  // Preload B fragments of Whh^T: wave w covers n-tiles {2w, 2w+1}, K=64 in 2 chunks.
  v16h bf[2][2];
  const int n_lane = lane & 15;
  const int kb_b   = (lane < 16) ? 0 : 8;
#pragma unroll
  for (int j = 0; j < 2; ++j) {
    int n = (wave * 2 + j) * 16 + n_lane;
    const float* __restrict__ wrow = Whh + (size_t)n * HID;   // B[k][n] = Whh[n][k]
#pragma unroll
    for (int kc = 0; kc < 2; ++kc) {
      int k0 = kc * 32;
      bf[j][kc] = cat16(cvt8(wrow + k0 + kb_b), cvt8(wrow + k0 + kb_b + 16));
    }
  }
  __syncthreads();

  const int m_a  = lane & 15;
  const int kb_a = (lane < 16) ? 0 : 8;

  for (int t = 0; t < T; ++t) {
    // Prefetch next step's xg rows (hides the only global latency in the chain).
    {
      int tn = (t + 1 < T) ? (t + 1) : t;
      const float* p0 = xg + (size_t)tn * G4 + wave * 32 + lane;
      __builtin_prefetch(p0, 0, 3);
      __builtin_prefetch(p0 + (size_t)T * G4, 0, 3);
    }
    // A fragments from current h: unconditional 128-bit LDS loads.
    const _Float16* hrow = &hf[m_a][0];
    v16h a0 = cat16(*(const v8h*)(hrow + kb_a),      *(const v8h*)(hrow + kb_a + 16));
    v16h a1 = cat16(*(const v8h*)(hrow + 32 + kb_a), *(const v8h*)(hrow + 32 + kb_a + 16));
#pragma unroll
    for (int j = 0; j < 2; ++j) {
      v8f acc = {};
      acc = __builtin_amdgcn_wmma_f32_16x16x32_f16(false, a0, false, bf[j][0],
                                                   (short)0, acc, false, false);
      acc = __builtin_amdgcn_wmma_f32_16x16x32_f16(false, a1, false, bf[j][1],
                                                   (short)0, acc, false, false);
      if (lane < 16) {                       // row m lives in acc[m] for lanes 0-15
        int n = (wave * 2 + j) * 16 + lane;
        gsh[0][n] = acc[0] + xg[(size_t)(0 * T + t) * G4 + n];
        gsh[1][n] = acc[1] + xg[(size_t)(1 * T + t) * G4 + n];
      }
    }
    __syncthreads();
    if (tid < BATCH * HID) {
      int b = tid >> 6, h = tid & 63;
      float ig = gsh[b][h], fg = gsh[b][HID + h];
      float gg = gsh[b][2 * HID + h], og = gsh[b][3 * HID + h];
      float c = sigf(fg) * cs[b][h] + sigf(ig) * tanhf(gg);
      float hv = sigf(og) * tanhf(c);
      cs[b][h] = c; hf[b][h] = (_Float16)hv;
      out[(size_t)(b * T + t) * HID + h] = hv;
    }
    __syncthreads();
  }
}

// ---------------------------------------------------------------------------
// Hidden-dim stats: Hc[B,3,T] = {max, mean, std(ddof=1)} over H[B,T,64].
// ---------------------------------------------------------------------------
__global__ void stats_kernel(const float* __restrict__ H, float* __restrict__ Hc, int T)
{
  int i = blockIdx.x * blockDim.x + threadIdx.x;
  if (i >= BATCH * T) return;
  int b = i / T, t = i % T;
  const float* p = H + (size_t)(b * T + t) * HID;
  float mx = -1e30f, s = 0.f, s2 = 0.f;
#pragma unroll 8
  for (int h = 0; h < HID; ++h) { float v = p[h]; mx = fmaxf(mx, v); s += v; s2 += v * v; }
  float mean = s * (1.0f / HID);
  float var_ub = (s2 - (float)HID * mean * mean) * (1.0f / (HID - 1));
  Hc[(b * 3 + 0) * T + t] = mx;
  Hc[(b * 3 + 1) * T + t] = mean;
  Hc[(b * 3 + 2) * T + t] = sqrtf(fmaxf(var_ub, 0.0f));
}

// ---------------------------------------------------------------------------
// Conv1d (K=11, pad 5) + BatchNorm(batch stats, biased var, eps 1e-5) [+ReLU].
// Single workgroup: conv -> shared reduction of per-channel sums -> normalize.
// ---------------------------------------------------------------------------
__global__ __launch_bounds__(1024) void conv_bn_kernel(
    const float* __restrict__ x, const float* __restrict__ w,
    const float* __restrict__ bias, float* __restrict__ tmp,
    float* __restrict__ y, int Cin, int Cout, int T, int relu)
{
  __shared__ float ssum[8], ssq[8];
  if (threadIdx.x < 8) { ssum[threadIdx.x] = 0.f; ssq[threadIdx.x] = 0.f; }
  __syncthreads();

  const int total = BATCH * Cout * T;
  for (int e = threadIdx.x; e < total; e += blockDim.x) {
    int t = e % T, o = (e / T) % Cout, b = e / (T * Cout);
    float acc = bias[o];
    for (int c = 0; c < Cin; ++c) {
      const float* xr = x + (size_t)(b * Cin + c) * T;
      const float* wr = w + (size_t)(o * Cin + c) * 11;
#pragma unroll
      for (int k = 0; k < 11; ++k) {
        int tt = t + k - 5;
        if (tt >= 0 && tt < T) acc += xr[tt] * wr[k];
      }
    }
    tmp[e] = acc;
    atomicAdd(&ssum[o], acc);
    atomicAdd(&ssq[o], acc * acc);
  }
  __syncthreads();

  const float invN = 1.0f / (float)(BATCH * T);
  for (int e = threadIdx.x; e < total; e += blockDim.x) {
    int o = (e / T) % Cout;
    float m = ssum[o] * invN;
    float v = ssq[o] * invN - m * m;
    float r = (tmp[e] - m) * rsqrtf(v + 1e-5f);
    if (relu) r = fmaxf(r, 0.0f);
    y[e] = r;
  }
}

// Hp[b,t,h] = H[b,t,h] + sigmoid(conv_out[b,0,t])
__global__ void gate_add_kernel(const float* __restrict__ H,
                                const float* __restrict__ xc,
                                float* __restrict__ Hp, int T)
{
  int i = blockIdx.x * blockDim.x + threadIdx.x;
  if (i >= BATCH * T * HID) return;
  int bt = i >> 6;
  int b = bt / T, t = bt % T;
  Hp[i] = H[i] + sigf(xc[b * T + t]);
}

// Collapse fc2(fc1(x)) (no nonlinearity between) into one 64->1 linear.
__global__ void fcprep_kernel(const float* __restrict__ fc1_w, const float* __restrict__ fc1_b,
                              const float* __restrict__ fc2_w, const float* __restrict__ fc2_b,
                              float* __restrict__ v)
{
  int h = threadIdx.x;
  if (h < HID) {
    float s = 0.f;
    for (int j = 0; j < HID; ++j) s += fc2_w[j] * fc1_w[j * HID + h];
    v[h] = s;
  }
  if (h == 0) {
    float bb = fc2_b[0];
    for (int j = 0; j < HID; ++j) bb += fc2_w[j] * fc1_b[j];
    v[HID] = bb;
  }
}

__global__ void final_kernel(const float* __restrict__ out2,
                             const float* __restrict__ v,
                             float* __restrict__ y, int T)
{
  int i = blockIdx.x * blockDim.x + threadIdx.x;
  if (i >= BATCH * T) return;
  const float* p = out2 + (size_t)i * HID;
  float s = v[HID];
#pragma unroll 8
  for (int h = 0; h < HID; ++h) s += p[h] * v[h];
  y[i] = sigf(s);
}

// ---------------------------------------------------------------------------
extern "C" void kernel_launch(void* const* d_in, const int* in_sizes, int n_in,
                              void* d_out, int out_size, void* d_ws, size_t ws_size,
                              hipStream_t stream)
{
  (void)in_sizes; (void)n_in; (void)out_size; (void)ws_size;
  const float* data    = (const float*)d_in[0];
  const float* h01     = (const float*)d_in[1];
  const float* c01     = (const float*)d_in[2];
  const float* h02     = (const float*)d_in[3];
  const float* c02     = (const float*)d_in[4];
  const float* l1_Wih0 = (const float*)d_in[5];
  const float* l1_Wih12= (const float*)d_in[6];
  const float* l1_Whh  = (const float*)d_in[7];
  const float* l1_b    = (const float*)d_in[8];
  const float* l2_Wih  = (const float*)d_in[9];
  const float* l2_Whh  = (const float*)d_in[10];
  const float* l2_b    = (const float*)d_in[11];
  const float* cw1 = (const float*)d_in[12]; const float* cb1 = (const float*)d_in[13];
  const float* cw2 = (const float*)d_in[14]; const float* cb2 = (const float*)d_in[15];
  const float* cw3 = (const float*)d_in[16]; const float* cb3 = (const float*)d_in[17];
  const float* cw4 = (const float*)d_in[18]; const float* cb4 = (const float*)d_in[19];
  const float* fc1_w = (const float*)d_in[20]; const float* fc1_b = (const float*)d_in[21];
  const float* fc2_w = (const float*)d_in[22]; const float* fc2_b = (const float*)d_in[23];
  float* out = (float*)d_out;

  const int T = T_LEN, M = BATCH * T_LEN;

  // Workspace layout (floats)
  float* ws    = (float*)d_ws;
  float* xg    = ws;                        // 8192*256
  float* bufA  = xg    + (size_t)M * G4;    // 8192*64
  float* bufB  = bufA  + (size_t)M * HID;
  float* Hkeep = bufB  + (size_t)M * HID;
  float* Xpad  = Hkeep + (size_t)M * HID;   // 8192*64
  float* Wpad  = Xpad  + (size_t)M * HID;   // 256*64
  float* Hc    = Wpad  + G4 * HID;          // 2*3*4096
  float* ct1   = Hc    + BATCH * 3 * T;     // up to 2*5*4096
  float* ct2   = ct1   + BATCH * 5 * T;
  float* ctmp  = ct2   + BATCH * 5 * T;
  float* vfc   = ctmp  + BATCH * 5 * T;     // 65

  const int pblocks = (M / 16) * 4 / 4;     // 2048 waves, 4 waves/block = 512 blocks
  dim3 pb(128);

  // ---- LSTM stack 1 ----
  pad_kernel<<<(M * HID + 255) / 256, 256, 0, stream>>>(data, Xpad, M, 40);
  pad_kernel<<<(G4 * HID + 255) / 256, 256, 0, stream>>>(l1_Wih0, Wpad, G4, 40);
  proj_gemm_kernel<<<pblocks, pb, 0, stream>>>(Xpad, Wpad, l1_b + 0 * G4, xg);
  lstm_scan_kernel<<<1, 256, 0, stream>>>(xg, l1_Whh + 0 * G4 * HID,
                                          h01 + 0 * BATCH * HID, c01 + 0 * BATCH * HID, bufA, T);
  proj_gemm_kernel<<<pblocks, pb, 0, stream>>>(bufA, l1_Wih12 + 0 * G4 * HID, l1_b + 1 * G4, xg);
  lstm_scan_kernel<<<1, 256, 0, stream>>>(xg, l1_Whh + 1 * G4 * HID,
                                          h01 + 1 * BATCH * HID, c01 + 1 * BATCH * HID, bufB, T);
  proj_gemm_kernel<<<pblocks, pb, 0, stream>>>(bufB, l1_Wih12 + 1 * G4 * HID, l1_b + 2 * G4, xg);
  lstm_scan_kernel<<<1, 256, 0, stream>>>(xg, l1_Whh + 2 * G4 * HID,
                                          h01 + 2 * BATCH * HID, c01 + 2 * BATCH * HID, Hkeep, T);

  // ---- temporal attention gate ----
  stats_kernel<<<(BATCH * T + 255) / 256, 256, 0, stream>>>(Hkeep, Hc, T);
  conv_bn_kernel<<<1, 1024, 0, stream>>>(Hc,  cw1, cb1, ctmp, ct1, 3, 3, T, 1);
  conv_bn_kernel<<<1, 1024, 0, stream>>>(ct1, cw2, cb2, ctmp, ct2, 3, 5, T, 1);
  conv_bn_kernel<<<1, 1024, 0, stream>>>(ct2, cw3, cb3, ctmp, ct1, 5, 5, T, 1);
  conv_bn_kernel<<<1, 1024, 0, stream>>>(ct1, cw4, cb4, ctmp, ct2, 5, 1, T, 0);
  gate_add_kernel<<<(BATCH * T * HID + 255) / 256, 256, 0, stream>>>(Hkeep, ct2, bufA, T);

  // ---- LSTM stack 2 ----
  proj_gemm_kernel<<<pblocks, pb, 0, stream>>>(bufA, l2_Wih + 0 * G4 * HID, l2_b + 0 * G4, xg);
  lstm_scan_kernel<<<1, 256, 0, stream>>>(xg, l2_Whh + 0 * G4 * HID,
                                          h02 + 0 * BATCH * HID, c02 + 0 * BATCH * HID, bufB, T);
  proj_gemm_kernel<<<pblocks, pb, 0, stream>>>(bufB, l2_Wih + 1 * G4 * HID, l2_b + 1 * G4, xg);
  lstm_scan_kernel<<<1, 256, 0, stream>>>(xg, l2_Whh + 1 * G4 * HID,
                                          h02 + 1 * BATCH * HID, c02 + 1 * BATCH * HID, bufA, T);
  proj_gemm_kernel<<<pblocks, pb, 0, stream>>>(bufA, l2_Wih + 2 * G4 * HID, l2_b + 2 * G4, xg);
  lstm_scan_kernel<<<1, 256, 0, stream>>>(xg, l2_Whh + 2 * G4 * HID,
                                          h02 + 2 * BATCH * HID, c02 + 2 * BATCH * HID, bufB, T);

  // ---- fused FC head ----
  fcprep_kernel<<<1, 64, 0, stream>>>(fc1_w, fc1_b, fc2_w, fc2_b, vfc);
  final_kernel<<<(BATCH * T + 255) / 256, 256, 0, stream>>>(bufB, vfc, out, T);
}
